// TSGR_84138409328991
// MI455X (gfx1250) — matrix operand order repaired and checked by reading, not compiled, not run on hardware
//
// CDNA5 / gfx1250 (MI455X) implementation of the masked graph-transformer autoencoder.
// Roofline reasoning (compile-only, from MI455X specs):
//  - The two adj (16384x16384 fp32, 1 GB) GEMMs are memory-bound: 68.7 GFLOP over
//    1 GB = 64 FLOP/byte vs 23.3 TB/s HBM. So the design goal is to stream adj
//    EXACTLY ONCE per GEMM: 64x128 C-tiles (grid.y = 1 for D=128 outputs), A-tile
//    staged into LDS via GLOBAL_LOAD_ASYNC_TO_LDS_B128 (ASYNCcnt), fp32->bf16
//    conversion folded into fragment build, and all matrix math on
//    v_wmma_f32_16x16x32_bf16 (f32 accumulate).
//  - Flash-style attention (online softmax) so the 1024x1024 score matrices are
//    never materialized; 4 WMMAs per 32-key chunk per wave.
//  - argsort/masking collapses algebraically to a per-token rank test.
#include <hip/hip_runtime.h>
#include <math.h>

#define NTOK   1024
#define NBATCH 16
#define NTOT   (NBATCH * NTOK)   // 16384
#define DMODEL 128
#define FFDIM  512
#define NHEAD  4
#define DHEAD  32

typedef __attribute__((ext_vector_type(16))) __bf16 bf16x16;
typedef __attribute__((ext_vector_type(8)))  __bf16 bf16x8;
typedef __attribute__((ext_vector_type(8)))  float  f32x8;
typedef int v4i __attribute__((vector_size(16)));  // matches builtin param type

#if __has_builtin(__builtin_amdgcn_global_load_async_to_lds_b128)
#define HAVE_ASYNC_LDS 1
#else
#define HAVE_ASYNC_LDS 0
#endif

// Builtin signature (from hipcc diagnostic): param0 = global (AS1) v4i*,
// param1 = LDS (AS3) v4i*, then offset + cpol immediates.
#define AS1V(p) ((__attribute__((address_space(1))) v4i*)(p))
#define AS3V(p) ((__attribute__((address_space(3))) v4i*)(p))

__device__ __forceinline__ void wait_async0() {
#if __has_builtin(__builtin_amdgcn_s_wait_asynccnt)
  __builtin_amdgcn_s_wait_asynccnt(0);
#else
  asm volatile("s_wait_asynccnt 0" ::: "memory");
#endif
}

// ---------------------------------------------------------------------------
// WMMA helpers: hand-built register fragments matching the CDNA5 ISA layouts
// (CDNA5 ISA §7.12.2), wave32.
// ---------------------------------------------------------------------------
__device__ __forceinline__ f32x8 wmma_bf16(bf16x16 a, bf16x16 b, f32x8 c) {
  return __builtin_amdgcn_wmma_f32_16x16x32_bf16(false, a, false, b, (short)0, c,
                                                 false, false);
}

// A-matrix 16x32 bf16 fragment from a bf16 LDS row (attention path).
// Element e -> K = half*8+e (e<8) / 16+half*8+(e-8).
__device__ __forceinline__ bf16x16 load_a_frag(const __bf16* row, int half) {
  bf16x8 lo = *(const bf16x8*)(row + half * 8);
  bf16x8 hi = *(const bf16x8*)(row + 16 + half * 8);
  bf16x16 a;
#pragma unroll
  for (int i = 0; i < 8; i++) { a[i] = lo[i]; a[i + 8] = hi[i]; }
  return a;
}

// A-matrix 16x32 fragment built from an fp32 LDS row (async-staged GEMM path):
// two 32-byte ds_load_b128 pairs + v_cvt_pk_bf16_f32 conversion in-register.
__device__ __forceinline__ bf16x16 load_a_frag_f32(const float* row, int half) {
  const float4 f0 = *(const float4*)(row + half * 8);
  const float4 f1 = *(const float4*)(row + half * 8 + 4);
  const float4 f2 = *(const float4*)(row + 16 + half * 8);
  const float4 f3 = *(const float4*)(row + 16 + half * 8 + 4);
  bf16x16 a;
  a[0] = (__bf16)f0.x; a[1] = (__bf16)f0.y; a[2] = (__bf16)f0.z; a[3] = (__bf16)f0.w;
  a[4] = (__bf16)f1.x; a[5] = (__bf16)f1.y; a[6] = (__bf16)f1.z; a[7] = (__bf16)f1.w;
  a[8]  = (__bf16)f2.x; a[9]  = (__bf16)f2.y; a[10] = (__bf16)f2.z; a[11] = (__bf16)f2.w;
  a[12] = (__bf16)f3.x; a[13] = (__bf16)f3.y; a[14] = (__bf16)f3.z; a[15] = (__bf16)f3.w;
  return a;
}

// B-matrix 32x16 bf16 fragment: lane's n-column stored as a k-contiguous LDS
// row. Element e -> K = half*8+e (e<8) / 16+half*8+(e-8).
__device__ __forceinline__ bf16x16 load_b_frag(const __bf16* row, int half) {
  bf16x8 lo = *(const bf16x8*)(row + half * 8);
  bf16x8 hi = *(const bf16x8*)(row + 16 + half * 8);
  bf16x16 b;
#pragma unroll
  for (int i = 0; i < 8; i++) { b[i] = lo[i]; b[i + 8] = hi[i]; }
  return b;
}

// ---------------------------------------------------------------------------
// Tiled GEMM:  C[M,Nn] = A[M,K] * B  (+bias[n]) (+resid[M,Nn]) (+relu)
//   BT=true : B is (Nn x K) row-major  (the "x @ W.T" pattern)
//   BT=false: B is (K x Nn) row-major  (adj @ X, s @ W)
// Block = 256 threads (8 waves), C-tile 64x128, K-step 32. Each wave owns a
// 16x64 strip: one A-fragment reused across 4 WMMAs per K-step.
// A tile staged fp32 (async LDS-DMA when available); B staged bf16.
// M%64==0, Nn%128==0, K%32==0 hold for every call site.
// ---------------------------------------------------------------------------
template <bool BT>
__global__ __launch_bounds__(256) void gemm_kernel(
    const float* __restrict__ A, const float* __restrict__ Bm,
    const float* __restrict__ bias, const float* __restrict__ resid,
    float* __restrict__ C, int M, int Nn, int K, int lda, int ldb, int ldc,
    int relu) {
  __shared__ __align__(16) float  Asf[64][36];   // [m][k] fp32, +4 pad
  __shared__ __align__(16) __bf16 Bs[128][40];   // [n][k] bf16, +8 pad
  const int mBase = blockIdx.x * 64;
  const int nBase = blockIdx.y * 128;
  const int t = threadIdx.x;
  const int w = t >> 5, L = t & 31, half = L >> 4, lm = L & 15;
  const int wr = w & 3, ch = w >> 2;  // wave -> 16-row strip, 64-col half
  f32x8 acc[4];
#pragma unroll
  for (int s2 = 0; s2 < 4; s2++) acc[s2] = (f32x8){0, 0, 0, 0, 0, 0, 0, 0};

  for (int kb = 0; kb < K; kb += 32) {
    __syncthreads();
    {  // ---- stage A tile (64 x 32 fp32) ----
      const int r = t >> 3;        // 0..31
      const int c = (t & 7) * 4;   // 0..28
#if HAVE_ASYNC_LDS
#pragma unroll
      for (int rr = 0; rr < 64; rr += 32) {
        float* gsrc = const_cast<float*>(A) + (size_t)(mBase + r + rr) * lda + kb + c;
        __builtin_amdgcn_global_load_async_to_lds_b128(
            AS1V(gsrc), AS3V(&Asf[r + rr][c]), 0, 0);
      }
#else
#pragma unroll
      for (int rr = 0; rr < 64; rr += 32) {
        const float* gsrc = A + (size_t)(mBase + r + rr) * lda + kb + c;
        *(float4*)&Asf[r + rr][c] = *(const float4*)gsrc;
        if (kb + 32 < K) __builtin_prefetch(gsrc + 32, 0, 1);
      }
#endif
      // ---- stage B tile as bf16 ----
      if (BT) {  // B is (Nn x K): tile rows n, cols k
#pragma unroll
        for (int rr = 0; rr < 128; rr += 32) {
          float4 v = *(const float4*)(Bm + (size_t)(nBase + r + rr) * ldb + kb + c);
          Bs[r + rr][c + 0] = (__bf16)v.x; Bs[r + rr][c + 1] = (__bf16)v.y;
          Bs[r + rr][c + 2] = (__bf16)v.z; Bs[r + rr][c + 3] = (__bf16)v.w;
        }
      } else {   // B is (K x Nn): tile rows k, cols n; store transposed
        const int kr = t >> 5;         // 0..7
        const int nc = (t & 31) * 4;   // 0..124
#pragma unroll
        for (int kk = 0; kk < 32; kk += 8) {
          float4 v = *(const float4*)(Bm + (size_t)(kb + kr + kk) * ldb + nBase + nc);
          Bs[nc + 0][kr + kk] = (__bf16)v.x; Bs[nc + 1][kr + kk] = (__bf16)v.y;
          Bs[nc + 2][kr + kk] = (__bf16)v.z; Bs[nc + 3][kr + kk] = (__bf16)v.w;
        }
      }
    }
#if HAVE_ASYNC_LDS
    wait_async0();   // our wave's async LDS-DMA writes landed
#endif
    __syncthreads();

    bf16x16 a = load_a_frag_f32(&Asf[wr * 16 + lm][0], half);
#pragma unroll
    for (int s2 = 0; s2 < 4; s2++) {
      bf16x16 b = load_b_frag(&Bs[ch * 64 + s2 * 16 + lm][0], half);
      acc[s2] = wmma_bf16(a, b, acc[s2]);
    }
  }

  const int colBase = nBase + ch * 64 + lm;
  float bvv[4];
#pragma unroll
  for (int s2 = 0; s2 < 4; s2++) bvv[s2] = bias ? bias[colBase + s2 * 16] : 0.f;
#pragma unroll
  for (int s2 = 0; s2 < 4; s2++) {
    const int col = colBase + s2 * 16;
#pragma unroll
    for (int v = 0; v < 8; v++) {
      const size_t row = (size_t)(mBase + wr * 16 + half * 8 + v);
      float val = acc[s2][v] + bvv[s2];
      if (resid) val += resid[row * ldc + col];
      if (relu) val = fmaxf(val, 0.f);
      C[row * ldc + col] = val;
    }
  }
}

// ---------------------------------------------------------------------------
// Flash attention: S=1024, dh=32, scale 1/sqrt(32). grid=(B*NH, S/128),
// block=256 (8 waves, 16 q-rows each). Q fragment is one 16x32 A-tile (K=dh).
// Per 32-key chunk: 2 WMMAs for Q*K^T, online-softmax, P staged through
// per-wave LDS to re-shape C-layout -> A-layout, 2 WMMAs for P*V.
// ---------------------------------------------------------------------------
__global__ __launch_bounds__(256) void attn_kernel(const float* __restrict__ qkv,
                                                   float* __restrict__ o) {
  __shared__ __align__(16) __bf16 Qs[128][40];      // [q][dh]
  __shared__ __align__(16) __bf16 Ks[32][40];       // [key][dh]
  __shared__ __align__(16) __bf16 Vt[32][40];       // [dh][key] (transposed)
  __shared__ __align__(16) __bf16 Ps[8][16][40];    // per-wave P scratch [q][key]
  const int bh = blockIdx.x;
  const int b = bh >> 2, h = bh & 3;
  const int qBase = blockIdx.y * 128;
  const size_t rowStart = (size_t)b * NTOK;
  const int ld = 3 * DMODEL;  // 384
  const float* qp = qkv + rowStart * ld + h * DHEAD;
  const float* kp = qkv + rowStart * ld + DMODEL + h * DHEAD;
  const float* vp = qkv + rowStart * ld + 2 * DMODEL + h * DHEAD;
  const int t = threadIdx.x;

  {  // stage Q tile (128 x 32) as bf16
    const int r = t >> 3, c = (t & 7) * 4;
#pragma unroll
    for (int rr = 0; rr < 128; rr += 32) {
      float4 v = *(const float4*)(qp + (size_t)(qBase + r + rr) * ld + c);
      Qs[r + rr][c + 0] = (__bf16)v.x; Qs[r + rr][c + 1] = (__bf16)v.y;
      Qs[r + rr][c + 2] = (__bf16)v.z; Qs[r + rr][c + 3] = (__bf16)v.w;
    }
  }
  __syncthreads();

  const int w = t >> 5, L = t & 31, half = L >> 4, lm = L & 15;
  const bf16x16 aq = load_a_frag(&Qs[w * 16 + lm][0], half);

  f32x8 o0 = {0, 0, 0, 0, 0, 0, 0, 0};
  f32x8 o1 = {0, 0, 0, 0, 0, 0, 0, 0};
  float mi[8], li[8];
#pragma unroll
  for (int v = 0; v < 8; v++) { mi[v] = -INFINITY; li[v] = 0.f; }
  const float scale = 0.17677669529663687f;  // 1/sqrt(32)

  for (int kb = 0; kb < NTOK; kb += 32) {
    __syncthreads();
    {  // stage K chunk (row-major) and V chunk (transposed)
      const int r = t >> 3, c = (t & 7) * 4;
      float4 kv = *(const float4*)(kp + (size_t)(kb + r) * ld + c);
      Ks[r][c + 0] = (__bf16)kv.x; Ks[r][c + 1] = (__bf16)kv.y;
      Ks[r][c + 2] = (__bf16)kv.z; Ks[r][c + 3] = (__bf16)kv.w;
      float4 vv = *(const float4*)(vp + (size_t)(kb + r) * ld + c);
      Vt[c + 0][r] = (__bf16)vv.x; Vt[c + 1][r] = (__bf16)vv.y;
      Vt[c + 2][r] = (__bf16)vv.z; Vt[c + 3][r] = (__bf16)vv.w;
    }
    __syncthreads();

    bf16x16 bk0 = load_b_frag(&Ks[lm][0], half);
    bf16x16 bk1 = load_b_frag(&Ks[16 + lm][0], half);
    f32x8 z = {0, 0, 0, 0, 0, 0, 0, 0};
    f32x8 s0 = wmma_bf16(aq, bk0, z);
    f32x8 s1 = wmma_bf16(aq, bk1, z);

    // Online softmax. C-layout row = half*8+v spans 16 lanes -> reduce with
    // xor shuffles {1,2,4,8} (stays inside the 16-lane half).
#pragma unroll
    for (int v = 0; v < 8; v++) {
      float x0 = s0[v] * scale, x1 = s1[v] * scale;
      float mv = fmaxf(x0, x1);
#pragma unroll
      for (int off = 1; off < 16; off <<= 1) mv = fmaxf(mv, __shfl_xor(mv, off, 32));
      float mn = fmaxf(mi[v], mv);
      float corr = __expf(mi[v] - mn);
      float e0 = __expf(x0 - mn), e1 = __expf(x1 - mn);
      float sum = e0 + e1;
#pragma unroll
      for (int off = 1; off < 16; off <<= 1) sum += __shfl_xor(sum, off, 32);
      li[v] = li[v] * corr + sum;
      mi[v] = mn;
      o0[v] *= corr; o1[v] *= corr;
      Ps[w][half * 8 + v][lm] = (__bf16)e0;         // keys kb..kb+15
      Ps[w][half * 8 + v][16 + lm] = (__bf16)e1;    // keys kb+16..kb+31
    }
    // Same-wave LDS RAW: DS ops are in-order per wave; compiler inserts waits.
    bf16x16 ap  = load_a_frag(&Ps[w][lm][0], half);
    bf16x16 bv0 = load_b_frag(&Vt[lm][0], half);
    bf16x16 bv1 = load_b_frag(&Vt[16 + lm][0], half);
    o0 = wmma_bf16(ap, bv0, o0);
    o1 = wmma_bf16(ap, bv1, o1);
  }

#pragma unroll
  for (int v = 0; v < 8; v++) {
    const size_t row = rowStart + qBase + w * 16 + half * 8 + v;
    const float inv = 1.f / li[v];
    o[row * DMODEL + h * DHEAD + lm] = o0[v] * inv;
    o[row * DMODEL + h * DHEAD + 16 + lm] = o1[v] * inv;
  }
}

// ---------------------------------------------------------------------------
// LayerNorm over D=128, one wave per row (8 rows / 256-thread block).
// ---------------------------------------------------------------------------
__global__ __launch_bounds__(256) void ln_kernel(const float* __restrict__ x,
                                                 const float* __restrict__ g,
                                                 const float* __restrict__ b,
                                                 float* __restrict__ y) {
  const int w = threadIdx.x >> 5, L = threadIdx.x & 31;
  const size_t row = (size_t)blockIdx.x * 8 + w;
  float4 v = *(const float4*)(x + row * DMODEL + L * 4);
  float s = v.x + v.y + v.z + v.w;
  float q = v.x * v.x + v.y * v.y + v.z * v.z + v.w * v.w;
#pragma unroll
  for (int off = 1; off < 32; off <<= 1) { s += __shfl_xor(s, off, 32); q += __shfl_xor(q, off, 32); }
  const float mean = s * (1.f / DMODEL);
  const float var = q * (1.f / DMODEL) - mean * mean;
  const float rstd = rsqrtf(var + 1e-5f);
  float4 gv = *(const float4*)(g + L * 4);
  float4 bv = *(const float4*)(b + L * 4);
  float4 outv;
  outv.x = (v.x - mean) * rstd * gv.x + bv.x;
  outv.y = (v.y - mean) * rstd * gv.y + bv.y;
  outv.z = (v.z - mean) * rstd * gv.z + bv.z;
  outv.w = (v.w - mean) * rstd * gv.w + bv.w;
  *(float4*)(y + row * DMODEL + L * 4) = outv;
}

// raw[row] = dot(mid[row], lin_w) + lin_b. One wave per row.
__global__ __launch_bounds__(256) void rowdot_kernel(const float* __restrict__ x,
                                                     const float* __restrict__ wv,
                                                     const float* __restrict__ bb,
                                                     float* __restrict__ out) {
  const int w = threadIdx.x >> 5, L = threadIdx.x & 31;
  const size_t row = (size_t)blockIdx.x * 8 + w;
  float4 a = *(const float4*)(x + row * DMODEL + L * 4);
  float4 g = *(const float4*)(wv + L * 4);
  float s = a.x * g.x + a.y * g.y + a.z * g.z + a.w * g.w;
#pragma unroll
  for (int off = 1; off < 32; off <<= 1) s += __shfl_xor(s, off, 32);
  if (L == 0) out[row] = s + bb[0];
}

// softmax over the 1024 tokens of one batch; writes score buffer AND the
// second output region of d_out.
__global__ __launch_bounds__(256) void softmax_kernel(const float* __restrict__ raw,
                                                      float* __restrict__ score,
                                                      float* __restrict__ tail) {
  const int b = blockIdx.x, t = threadIdx.x;
  const int w = t >> 5, L = t & 31;
  const float* rb = raw + (size_t)b * NTOK;
  float v[4];
  float m = -INFINITY;
#pragma unroll
  for (int k = 0; k < 4; k++) { v[k] = rb[t + k * 256]; m = fmaxf(m, v[k]); }
#pragma unroll
  for (int off = 1; off < 32; off <<= 1) m = fmaxf(m, __shfl_xor(m, off, 32));
  __shared__ float sred[8];
  if (L == 0) sred[w] = m;
  __syncthreads();
  float M = -INFINITY;
#pragma unroll
  for (int i = 0; i < 8; i++) M = fmaxf(M, sred[i]);
  __syncthreads();
  float sum = 0.f;
#pragma unroll
  for (int k = 0; k < 4; k++) sum += __expf(v[k] - M);
#pragma unroll
  for (int off = 1; off < 32; off <<= 1) sum += __shfl_xor(sum, off, 32);
  if (L == 0) sred[w] = sum;
  __syncthreads();
  float total = 0.f;
#pragma unroll
  for (int i = 0; i < 8; i++) total += sred[i];
  const float inv = 1.f / total;
#pragma unroll
  for (int k = 0; k < 4; k++) {
    const float e = __expf(v[k] - M) * inv;
    score[(size_t)b * NTOK + t + k * 256] = e;
    tail[(size_t)b * NTOK + t + k * 256] = e;
  }
}

// Masking. The resort/gather in the reference is an inverse permutation, so:
// new_mid[b,t] = (ascRank(score[b,t]) >= mask_len ? mid[b,t] : mask_token) * score[b,t]
// Stable ascending rank = #(score< ) + #(score== with smaller index).
__global__ __launch_bounds__(128) void mask_kernel(const float* __restrict__ score,
                                                   const float* __restrict__ mid,
                                                   const float* __restrict__ mask_token,
                                                   const int* __restrict__ mask_len_p,
                                                   float* __restrict__ nm) {
  const int idx = blockIdx.x;
  const int b = idx >> 10, tk = idx & 1023;
  const float sv = score[(size_t)b * NTOK + tk];
  const int tI = threadIdx.x;
  int cnt = 0;
  for (int j = tI; j < NTOK; j += 128) {
    const float o = score[(size_t)b * NTOK + j];
    cnt += (o < sv) || (o == sv && j < tk);
  }
  __shared__ int sc[128];
  sc[tI] = cnt;
  __syncthreads();
  for (int s2 = 64; s2 > 0; s2 >>= 1) {
    if (tI < s2) sc[tI] += sc[tI + s2];
    __syncthreads();
  }
  const int rank = sc[0];
  const bool keep = rank >= mask_len_p[0];
  const size_t base = ((size_t)b * NTOK + tk) * DMODEL;
  const float val = keep ? mid[base + tI] : mask_token[tI];
  nm[base + tI] = val * sv;
}

// ---------------------------------------------------------------------------
// Host orchestration
// ---------------------------------------------------------------------------
extern "C" void kernel_launch(void* const* d_in, const int* in_sizes, int n_in,
                              void* d_out, int out_size, void* d_ws, size_t ws_size,
                              hipStream_t stream) {
  // setup_inputs() dict order:
  const float* X_concat  = (const float*)d_in[0];
  const float* adj       = (const float*)d_in[1];
  const float* w_in      = (const float*)d_in[2];
  const float* b_in      = (const float*)d_in[3];
  const float* gcn_enc_w = (const float*)d_in[4];
  const float* gcn_enc_b = (const float*)d_in[5];
  const float* lin_w     = (const float*)d_in[6];
  const float* lin_b     = (const float*)d_in[7];
  const float* mask_tok  = (const float*)d_in[8];
  const float* gcn_dec_w = (const float*)d_in[9];
  const float* gcn_dec_b = (const float*)d_in[10];
  const float* w_out     = (const float*)d_in[11];
  const float* b_out     = (const float*)d_in[12];
  // d_in[13] = num_tokens (==1024, compile-time), d_in[14] = mask_len
  const int*   mask_len  = (const int*)d_in[14];
  const float* const* P  = (const float* const*)d_in;  // layer params by index

  // Workspace carve-up (~126 MB fp32)
  size_t off = 0;
  float* wsf = (float*)d_ws;
  auto take = [&](size_t n) { float* p = wsf + off; off += n; return p; };
  float* xbuf = take((size_t)NTOT * DMODEL);
  float* qkvb = take((size_t)NTOT * 3 * DMODEL);
  float* obuf = take((size_t)NTOT * DMODEL);
  float* rbuf = take((size_t)NTOT * DMODEL);
  float* x1b  = take((size_t)NTOT * DMODEL);
  float* ffb  = take((size_t)NTOT * FFDIM);
  float* r2b  = take((size_t)NTOT * DMODEL);
  float* sbuf = take((size_t)NTOT * DMODEL);
  float* tbuf = take((size_t)NTOT * DMODEL);
  float* midb = take((size_t)NTOT * DMODEL);
  float* rawb = take((size_t)NTOT);
  float* scrb = take((size_t)NTOT);
  (void)ws_size; (void)n_in; (void)in_sizes; (void)out_size;

  const dim3 blk(256);
  auto gemm_nt = [&](const float* A, const float* B, const float* bias,
                     const float* resid, float* C, int M, int Nn, int K,
                     int relu) {
    dim3 g(M / 64, Nn / 128);
    gemm_kernel<true><<<g, blk, 0, stream>>>(A, B, bias, resid, C, M, Nn, K, K, K, Nn, relu);
  };
  auto gemm_nn = [&](const float* A, const float* B, const float* bias,
                     float* C, int M, int Nn, int K, int lda, int relu) {
    dim3 g(M / 64, Nn / 128);
    gemm_kernel<false><<<g, blk, 0, stream>>>(A, B, bias, nullptr, C, M, Nn, K, lda, Nn, Nn, relu);
  };
  auto layernorm = [&](const float* x, const float* g, const float* b, float* y) {
    ln_kernel<<<NTOT / 8, blk, 0, stream>>>(x, g, b, y);
  };
  // Transformer layer: params at base index i0 = {wqkv,bqkv,wo,bo,ln1g,ln1b,
  // wf1,bf1,wf2,bf2,ln2g,ln2b}
  auto tlayer = [&](const float* xin, int i0, float* xout) {
    gemm_nt(xin, P[i0 + 0], P[i0 + 1], nullptr, qkvb, NTOT, 3 * DMODEL, DMODEL, 0);
    attn_kernel<<<dim3(NBATCH * NHEAD, NTOK / 128), blk, 0, stream>>>(qkvb, obuf);
    gemm_nt(obuf, P[i0 + 2], P[i0 + 3], xin, rbuf, NTOT, DMODEL, DMODEL, 0);
    layernorm(rbuf, P[i0 + 4], P[i0 + 5], x1b);
    gemm_nt(x1b, P[i0 + 6], P[i0 + 7], nullptr, ffb, NTOT, FFDIM, DMODEL, 1);
    gemm_nt(ffb, P[i0 + 8], P[i0 + 9], x1b, r2b, NTOT, DMODEL, FFDIM, 0);
    layernorm(r2b, P[i0 + 10], P[i0 + 11], xout);
  };

  // 1. input projection
  gemm_nt(X_concat, w_in, b_in, nullptr, xbuf, NTOT, DMODEL, DMODEL, 0);
  // 2. encoder transformer layer (enc params start at index 15)
  tlayer(xbuf, 15, sbuf);
  // 3-4. encoder GCN: mid = relu(adj @ (s @ gcn_enc_w) + gcn_enc_b)
  gemm_nn(sbuf, gcn_enc_w, nullptr, tbuf, NTOT, DMODEL, DMODEL, DMODEL, 0);
  gemm_nn(adj, tbuf, gcn_enc_b, midb, NTOT, DMODEL, NTOT, NTOT, 1);
  // 5-6. token scores + per-batch softmax (also writes d_out tail)
  rowdot_kernel<<<NTOT / 8, blk, 0, stream>>>(midb, lin_w, lin_b, rawb);
  softmax_kernel<<<NBATCH, blk, 0, stream>>>(
      rawb, scrb, (float*)d_out + (size_t)NTOT * DMODEL);
  // 7. rank-based masking -> decoder input
  mask_kernel<<<NTOT, dim3(128), 0, stream>>>(scrb, midb, mask_tok, mask_len, xbuf);
  // 8. decoder transformer layer (dec params start at index 27)
  tlayer(xbuf, 27, sbuf);
  // 9-10. decoder GCN
  gemm_nn(sbuf, gcn_dec_w, nullptr, tbuf, NTOT, DMODEL, DMODEL, DMODEL, 0);
  gemm_nn(adj, tbuf, gcn_dec_b, midb, NTOT, DMODEL, NTOT, NTOT, 1);
  // 11. output projection straight into d_out
  gemm_nt(midb, w_out, b_out, nullptr, (float*)d_out, NTOT, DMODEL, DMODEL, 0);
}
